// MultiSpectralAttentionLayer_26139170964015
// MI455X (gfx1250) — matrix operand order repaired and checked by reading, not compiled, not run on hardware
//
#include <hip/hip_runtime.h>
#include <hip/hip_bf16.h>
#include <math.h>

// ---------------------------------------------------------------------------
// MultiSpectralAttentionLayer for MI455X (gfx1250, wave32)
// x:(32,512,56,56) f32. Memory-bound: 2 passes over x (~616MB -> ~26us @23.3TB/s).
// Small GEMMs (32x512x32 and 32x32x512) mapped to v_wmma_f32_16x16x32_f16.
// ---------------------------------------------------------------------------

typedef __attribute__((ext_vector_type(16))) _Float16 v16h;
typedef __attribute__((ext_vector_type(8)))  float    v8f;

#define N_BATCH 32
#define C_TOT   512
#define NSPLIT  16
#define CPART   32          // 512/16
#define IMG_HW  3136        // 56*56
#define POOLED  49          // 7*7

__device__ __constant__ int d_MX[16] = {0,0,6,0,0,1,1,4,5,1,3,0,0,0,3,2};
__device__ __constant__ int d_MY[16] = {0,1,0,5,2,0,2,0,0,6,0,4,6,3,5,2};

__device__ __forceinline__ float hard_sigmoid_f(float x) {
  return fminf(fmaxf((x + 3.0f) * (1.0f / 6.0f), 0.0f), 1.0f);
}

// One row of the 7x7 DCT-II table: D[f][p] = cos(pi*f*(p+0.5)/7)/sqrt(7), *sqrt(2) for f>0
__device__ __forceinline__ void dct_row(int f, float* out) {
  const float inv7  = 0.37796447300922720f;   // 1/sqrt(7)
  const float scale = (f == 0) ? inv7 : inv7 * 1.41421356237309515f;
  const float k = 3.14159265358979323846f * (float)f / 7.0f;
  #pragma unroll
  for (int p = 0; p < 7; ++p)
    out[p] = cosf(k * ((float)p + 0.5f)) * scale;
}

// ---------------------------------------------------------------------------
// Kernel 1: adaptive avg pool 56x56 -> 7x7 (exact 8x8 block means).
// One 256-thread block per (n,c) image. 128-bit staging into LDS (coalesced
// global_load_b128 / ds_store_b128), then 49 threads reduce their 8x8 block
// with ds_load_b128 (each 8-float row is 32B-aligned since 56 = 8*7).
// ---------------------------------------------------------------------------
__global__ void pool_kernel(const float* __restrict__ x, float* __restrict__ xp) {
  __shared__ float4 xs4[IMG_HW / 4];                // 784 float4 = 12.25 KB
  const int img = blockIdx.x;                       // n*512 + c
  const float4* src = (const float4*)(x + (size_t)img * IMG_HW);
  for (int i = threadIdx.x; i < IMG_HW / 4; i += 256) xs4[i] = src[i];
  __syncthreads();
  const int t = threadIdx.x;
  if (t < POOLED) {
    const int py = t / 7, px = t % 7;
    const float4* base = xs4 + py * 112 + px * 2;   // (py*8*56 + px*8)/4
    float s = 0.0f;
    #pragma unroll
    for (int dy = 0; dy < 8; ++dy) {
      const float4 r0 = base[dy * 14];              // 56/4 float4 per row
      const float4 r1 = base[dy * 14 + 1];
      s += (r0.x + r0.y + r0.z + r0.w) + (r1.x + r1.y + r1.z + r1.w);
    }
    xp[(size_t)img * POOLED + t] = s * (1.0f / 64.0f);
  }
}

// ---------------------------------------------------------------------------
// Kernel 2: grouped conv1 (7x7 -> 3x3, stride2) + hard_sigmoid, depthwise
// conv2 (3x3 -> 1x1, stride2), pred = clip(v*16, 0, 14.9) -> left/frac.
// One thread per (n, split): 512 threads total.
// ---------------------------------------------------------------------------
__global__ void conv_kernel(const float* __restrict__ xp,
                            const float* __restrict__ c1w, const float* __restrict__ c1b,
                            const float* __restrict__ c2w, const float* __restrict__ c2b,
                            int* __restrict__ left_out, float* __restrict__ frac_out) {
  const int t  = threadIdx.x;        // 0..511
  const int n  = t >> 4;
  const int sp = t & 15;
  const float* base = xp + ((size_t)n * C_TOT + sp * CPART) * POOLED;
  const float* wgt  = c1w + sp * CPART * 9;

  float a1[3][3];
  #pragma unroll
  for (int oy = 0; oy < 3; ++oy) {
    #pragma unroll
    for (int ox = 0; ox < 3; ++ox) {
      float acc = c1b[sp];
      for (int ic = 0; ic < CPART; ++ic) {
        const float* im = base + ic * POOLED;
        const float* wk = wgt + ic * 9;
        #pragma unroll
        for (int ky = 0; ky < 3; ++ky)
          #pragma unroll
          for (int kx = 0; kx < 3; ++kx)
            acc += im[(2 * oy + ky) * 7 + (2 * ox + kx)] * wk[ky * 3 + kx];
      }
      a1[oy][ox] = hard_sigmoid_f(acc);
    }
  }
  float v = c2b[sp];
  #pragma unroll
  for (int ky = 0; ky < 3; ++ky)
    #pragma unroll
    for (int kx = 0; kx < 3; ++kx)
      v += a1[ky][kx] * c2w[sp * 9 + ky * 3 + kx];

  float pred = fminf(fmaxf(v * (float)NSPLIT, 0.0f), (float)NSPLIT - 1.1f);
  int   l    = (int)pred;            // pred >= 0 -> truncation == floor
  left_out[t] = l;
  frac_out[t] = pred - (float)l;
}

// ---------------------------------------------------------------------------
// Kernel 3: per-channel separable DCT projection, blended between left/right
// frequency filters.  One thread per (n, c): 16384 threads.
// ---------------------------------------------------------------------------
__global__ void dct_kernel(const float* __restrict__ xp,
                           const int* __restrict__ left, const float* __restrict__ frac,
                           float* __restrict__ s_mat) {
  const int id = blockIdx.x * 256 + threadIdx.x;    // 0..16383
  const int n  = id >> 9;
  const int cc = id & 511;
  const int sp = cc >> 5;
  const int l  = left[n * NSPLIT + sp];
  const float f = frac[n * NSPLIT + sp];

  float dxl[7], dyl[7], dxr[7], dyr[7];
  dct_row(d_MX[l],     dxl);
  dct_row(d_MY[l],     dyl);
  dct_row(d_MX[l + 1], dxr);
  dct_row(d_MY[l + 1], dyr);

  const float* im = xp + (size_t)id * POOLED;
  float accl = 0.0f, accr = 0.0f;
  #pragma unroll
  for (int h = 0; h < 7; ++h) {
    float rl = 0.0f, rr = 0.0f;
    #pragma unroll
    for (int w = 0; w < 7; ++w) {
      const float v = im[h * 7 + w];
      rl += v * dyl[w];
      rr += v * dyr[w];
    }
    accl += dxl[h] * rl;
    accr += dxr[h] * rr;
  }
  s_mat[id] = accl * (1.0f - f) + accr * f;
}

// A-fragment K index (16-bit A 16x32 layout, ISA 7.12.2)
__device__ __forceinline__ int a_kidx(int v, int half) {
  return (v < 4) ? (8 * half + 2 * v) : (16 + 8 * half + 2 * (v - 4));
}
// B-fragment K index (same K order as A: lanes 0-15 hold K=0..7,16..23; 16-31 hold K=8..15,24..31)
__device__ __forceinline__ int b_kidx(int v, int half) {
  return (v < 4) ? (8 * half + 2 * v) : (16 + 8 * half + 2 * (v - 4));
}

// ---------------------------------------------------------------------------
// Kernel 4: hmid = hard_sigmoid(s @ w1^T)   (32x512)*(512x32) -> 32x32
// 4 waves, one 16x16 tile each; K=512 in 16 WMMA steps of 32.
// ---------------------------------------------------------------------------
__global__ void mm1_kernel(const float* __restrict__ s_mat,
                           const float* __restrict__ w1,
                           float* __restrict__ hmid) {
  const int wave = threadIdx.x >> 5;
  const int lane = threadIdx.x & 31;
  const int tm = wave >> 1, tj = wave & 1;
  const int half = lane >> 4, l16 = lane & 15;
  const int mrow = tm * 16 + l16;                   // A row (batch n)
  const int ncol = tj * 16 + l16;                   // B col (hidden j)

  v8f c = {};
  for (int K0 = 0; K0 < 512; K0 += 32) {
    v16h a, b;
    #pragma unroll
    for (int v = 0; v < 8; ++v) {
      const int ka = K0 + a_kidx(v, half);
      a[2 * v]     = (_Float16)s_mat[mrow * 512 + ka];
      a[2 * v + 1] = (_Float16)s_mat[mrow * 512 + ka + 1];
      const int kb = K0 + b_kidx(v, half);
      b[2 * v]     = (_Float16)w1[ncol * 512 + kb];
      b[2 * v + 1] = (_Float16)w1[ncol * 512 + kb + 1];
    }
    c = __builtin_amdgcn_wmma_f32_16x16x32_f16(false, a, false, b, (short)0, c,
                                               false, false);
  }
  #pragma unroll
  for (int j = 0; j < 8; ++j) {
    const int mo = tm * 16 + j + 8 * half;          // C/D layout: M = j + 8*(lane/16)
    hmid[mo * 32 + ncol] = hard_sigmoid_f(c[j]);
  }
}

// ---------------------------------------------------------------------------
// Kernel 5: cw = sigmoid(hmid @ w2^T)   (32x32)*(32x512) -> 32x512
// 64 single-wave blocks, one 16x16 tile each, K=32 (single WMMA).
// ---------------------------------------------------------------------------
__global__ void mm2_kernel(const float* __restrict__ hmid,
                           const float* __restrict__ w2,
                           float* __restrict__ cw) {
  const int lane = threadIdx.x;                     // blockDim == 32
  const int tm = blockIdx.x & 1, tn = blockIdx.x >> 1;
  const int half = lane >> 4, l16 = lane & 15;
  const int mrow = tm * 16 + l16;                   // batch n
  const int ccol = tn * 16 + l16;                   // channel c

  v16h a, b;
  #pragma unroll
  for (int v = 0; v < 8; ++v) {
    const int ka = a_kidx(v, half);
    a[2 * v]     = (_Float16)hmid[mrow * 32 + ka];
    a[2 * v + 1] = (_Float16)hmid[mrow * 32 + ka + 1];
    const int kb = b_kidx(v, half);
    b[2 * v]     = (_Float16)w2[ccol * 32 + kb];
    b[2 * v + 1] = (_Float16)w2[ccol * 32 + kb + 1];
  }
  v8f c = {};
  c = __builtin_amdgcn_wmma_f32_16x16x32_f16(false, a, false, b, (short)0, c,
                                             false, false);
  #pragma unroll
  for (int j = 0; j < 8; ++j) {
    const int mo = tm * 16 + j + 8 * half;
    cw[mo * 512 + ccol] = 1.0f / (1.0f + __expf(-c[j]));
  }
}

// ---------------------------------------------------------------------------
// Kernel 6: out = x * cw[n,c]  (bandwidth-critical pass, float4 vectorized)
// ---------------------------------------------------------------------------
__global__ void scale_kernel(const float* __restrict__ x,
                             const float* __restrict__ cw,
                             float* __restrict__ out) {
  const size_t i = (size_t)blockIdx.x * 256 + threadIdx.x;  // float4 index
  const int img = (int)(i / 784);                           // 3136/4 f4 per (n,c)
  const float s = cw[img];
  const float4 v = ((const float4*)x)[i];
  float4 r;
  r.x = v.x * s; r.y = v.y * s; r.z = v.z * s; r.w = v.w * s;
  ((float4*)out)[i] = r;
}

// ---------------------------------------------------------------------------
extern "C" void kernel_launch(void* const* d_in, const int* in_sizes, int n_in,
                              void* d_out, int out_size, void* d_ws, size_t ws_size,
                              hipStream_t stream) {
  const float* x   = (const float*)d_in[0];
  const float* c1w = (const float*)d_in[1];
  const float* c1b = (const float*)d_in[2];
  const float* c2w = (const float*)d_in[3];
  const float* c2b = (const float*)d_in[4];
  const float* w1  = (const float*)d_in[5];
  const float* w2  = (const float*)d_in[6];
  float* out = (float*)d_out;

  // Workspace layout (floats); total ~3.2 MB
  float* ws    = (float*)d_ws;
  float* xp    = ws;                  // 32*512*49 = 802816
  float* s_mat = ws + 802816;         // 32*512   = 16384
  float* hmid  = ws + 819200;         // 32*32    = 1024
  float* cw    = ws + 820224;         // 32*512   = 16384
  float* frac  = ws + 836608;         // 32*16    = 512
  int*   left  = (int*)(ws + 837120); // 32*16    = 512

  pool_kernel <<<N_BATCH * C_TOT, 256, 0, stream>>>(x, xp);
  conv_kernel <<<1, N_BATCH * NSPLIT, 0, stream>>>(xp, c1w, c1b, c2w, c2b, left, frac);
  dct_kernel  <<<(N_BATCH * C_TOT) / 256, 256, 0, stream>>>(xp, left, frac, s_mat);
  mm1_kernel  <<<1, 128, 0, stream>>>(s_mat, w1, hmid);
  mm2_kernel  <<<64, 32, 0, stream>>>(hmid, w2, cw);
  scale_kernel<<<(N_BATCH * C_TOT * IMG_HW) / (4 * 256), 256, 0, stream>>>(x, cw, out);
}
